// AttentionModel_87462714015827
// MI455X (gfx1250) — matrix operand order repaired and checked by reading, not compiled, not run on hardware
//
#include <hip/hip_runtime.h>
#include <math.h>

// ---------------- model dims ----------------
#define BB   16
#define SS   1024
#define DD   128
#define HH   8
#define DHH  16
#define LL   3
#define DFF  512
#define NC   6
#define BS   (BB * SS)          // 16384 rows

typedef __attribute__((ext_vector_type(16))) __bf16 v16bf;
typedef __attribute__((ext_vector_type(8)))  float  v8f;

// ---------------- WMMA fragment loaders (CDNA5 ISA 7.12.2) -----
// A fragment: 16x32 bf16, row-major source, ld = K stride.
// lane m = lane&15 is the row; half = lane>>4 selects K sub-block:
//   K(j) = 16*half + j
__device__ __forceinline__ v16bf load_a_frag(const __bf16* tile, int ld) {
    const int lane = threadIdx.x & 31;
    const int m = lane & 15, half = lane >> 4;
    const __bf16* row = tile + (size_t)m * ld;
    v16bf a;
#pragma unroll
    for (int j = 0; j < 16; ++j)
        a[j] = row[16 * half + j];
    return a;
}

// B fragment: 32x16 bf16 from TRANSPOSED storage Bt[N][K], ld = K stride.
// lane n = lane&15 is the column; K(j) = 16*half + j  (contiguous!)
__device__ __forceinline__ v16bf load_bt_frag(const __bf16* tileT, int ld) {
    const int lane = threadIdx.x & 31;
    const int n = lane & 15, half = lane >> 4;
    const __bf16* row = tileT + (size_t)n * ld + 16 * half;
    v16bf b;
#pragma unroll
    for (int j = 0; j < 16; ++j) b[j] = row[j];
    return b;
}

__device__ __forceinline__ v8f wmma_bf16(v16bf a, v16bf b, v8f c) {
    return __builtin_amdgcn_wmma_f32_16x16x32_bf16(
        false, a, false, b, (short)0, c, false, false);
}

// ---------------- weight convert + transpose:  dst[c*rows+r] = src[r*cols+c]
__global__ void k_transpose_bf16(const float* __restrict__ src,
                                 __bf16* __restrict__ dst, int rows, int cols) {
    int idx = blockIdx.x * blockDim.x + threadIdx.x;
    if (idx >= rows * cols) return;
    int r = idx / cols, c = idx % cols;
    dst[(size_t)c * rows + r] = (__bf16)src[(size_t)r * cols + c];
}

// copy x -> workspace fp32 master + bf16 shadow
__global__ void k_convert_x(const float* __restrict__ src,
                            float* __restrict__ x, __bf16* __restrict__ xb, int n) {
    int idx = blockIdx.x * blockDim.x + threadIdx.x;
    if (idx >= n) return;
    float v = src[idx];
    x[idx]  = v;
    xb[idx] = (__bf16)v;
}

// ---------------- fused QKV projection -----------------------------------
// grid (BS/64, H, 3); 1 wave computes a 64x16 tile (4 M-subtiles reuse the
// weight B-fragment). N-tile == head (DH==16). Double-buffered K loop.
// Q,K -> [b,h,s,dh] bf16 ; V -> [b,h,dh,s] bf16 (transposed for PV B-frag)
__global__ void k_qkv(const __bf16* __restrict__ xb,
                      const __bf16* __restrict__ wqt,  // [D][D] per layer, transposed
                      const __bf16* __restrict__ wkt,
                      const __bf16* __restrict__ wvt,
                      const float* __restrict__ bq,
                      const float* __restrict__ bk,
                      const float* __restrict__ bv,
                      __bf16* __restrict__ qo, __bf16* __restrict__ ko,
                      __bf16* __restrict__ vo) {
    const int tm = blockIdx.x, h = blockIdx.y, which = blockIdx.z;
    const int lane = threadIdx.x & 31;
    const int m = lane & 15, half = lane >> 4;

    const __bf16* wt   = which == 0 ? wqt : which == 1 ? wkt : wvt;
    const float*  bias = which == 0 ? bq  : which == 1 ? bk  : bv;

    const __bf16* Btile = wt + (size_t)(h * 16) * DD;
    const __bf16* Atile = xb + (size_t)(tm * 64) * DD;

    constexpr int NSTEP = DD / 32;          // 4
    v16bf bf[2], af[2][4];
    bf[0] = load_bt_frag(Btile, DD);
#pragma unroll
    for (int i = 0; i < 4; ++i) af[0][i] = load_a_frag(Atile + (size_t)(i * 16) * DD, DD);

    v8f acc[4] = {};
#pragma unroll
    for (int s = 0; s < NSTEP; ++s) {
        const int cur = s & 1, nxt = cur ^ 1;
        if (s + 1 < NSTEP) {
            bf[nxt] = load_bt_frag(Btile + (s + 1) * 32, DD);
#pragma unroll
            for (int i = 0; i < 4; ++i)
                af[nxt][i] = load_a_frag(Atile + (size_t)(i * 16) * DD + (s + 1) * 32, DD);
        }
#pragma unroll
        for (int i = 0; i < 4; ++i)
            acc[i] = wmma_bf16(af[cur][i], bf[cur], acc[i]);
    }
    const float bn = bias[h * 16 + m];   // column n = lane&15 = dh
#pragma unroll
    for (int i = 0; i < 4; ++i) {
#pragma unroll
        for (int r = 0; r < 8; ++r) {
            int gm = tm * 64 + i * 16 + r + 8 * half;    // global row
            int bI = gm / SS, s = gm % SS;
            float v = fmaxf(acc[i][r] + bn, 0.0f);       // ReLU
            __bf16 vb = (__bf16)v;
            size_t head = ((size_t)bI * HH + h);
            if (which == 0)      qo[(head * SS + s) * DHH + m] = vb;
            else if (which == 1) ko[(head * SS + s) * DHH + m] = vb;
            else                 vo[(head * DHH + m) * SS + s] = vb;
        }
    }
}

// ---------------- flash attention: grid (S/32, H, B), 1 wave/block --------
// 32 queries per wave (two 16-row A fragments share every K/V fragment).
__global__ void k_attn(const __bf16* __restrict__ Q,
                       const __bf16* __restrict__ Km,
                       const __bf16* __restrict__ Vt,
                       float* __restrict__ O) {
    __shared__ float  s_tile[32][32];
    __shared__ __bf16 p_tile[32][32];
    __shared__ float  s_m[32], s_l[32], s_corr[32];

    const int lane = threadIdx.x & 31;
    const int m = lane & 15, half = lane >> 4;
    const int qt = blockIdx.x, h = blockIdx.y, b = blockIdx.z;
    const float scale = 0.25f;                       // 1/sqrt(16)

    const size_t head = (size_t)b * HH + h;
    const __bf16* Qb = Q  + (head * SS + qt * 32) * DHH;
    const __bf16* Kb = Km + head * SS * DHH;
    const __bf16* Vb = Vt + head * DHH * SS;

    // Q A-fragments, DH=16 padded to K=32: K = 16*half+j, dh = j (half 1 meets zero B)
    v16bf aq0 = {}, aq1 = {};
    {
        const __bf16* qr0 = Qb + (size_t)m * DHH;
        const __bf16* qr1 = Qb + (size_t)(16 + m) * DHH;
#pragma unroll
        for (int j = 0; j < 16; ++j) { aq0[j] = qr0[j]; aq1[j] = qr1[j]; }
    }
    s_m[lane] = -3.0e38f;
    s_l[lane] = 0.0f;
    v8f acc0 = {}, acc1 = {};

    for (int kc = 0; kc < SS; kc += 32) {
        // prefetch next 32-key chunk of K and V (global_prefetch_b8)
        if (kc + 32 < SS) {
            __builtin_prefetch(Kb + (size_t)(kc + 32 + lane) * DHH, 0, 1);
            __builtin_prefetch(Vb + (size_t)m * SS + kc + 32 + 16 * half, 0, 1);
        }
        // ---- scores for 32 keys x 32 queries: four 16x16 WMMAs ----------
#pragma unroll
        for (int t = 0; t < 2; ++t) {
            v16bf bk = {};
            if (half == 0) {                          // dh = j (0..15); half==1 is pad
                const __bf16* kr = Kb + (size_t)(kc + t * 16 + m) * DHH;
#pragma unroll
                for (int j = 0; j < 16; ++j) bk[j] = kr[j];
            }
            v8f sc0 = {}, sc1 = {};
            sc0 = wmma_bf16(aq0, bk, sc0);
            sc1 = wmma_bf16(aq1, bk, sc1);
#pragma unroll
            for (int r = 0; r < 8; ++r) {
                s_tile[r + 8 * half][t * 16 + m]      = sc0[r];
                s_tile[16 + r + 8 * half][t * 16 + m] = sc1[r];
            }
        }
        __syncthreads();

        // ---- online softmax: one lane per query row (all 32 lanes) ------
        {
            float mold = s_m[lane], cmax = -3.0e38f;
#pragma unroll
            for (int j = 0; j < 32; ++j)
                cmax = fmaxf(cmax, s_tile[lane][j] * scale);
            float mnew = fmaxf(mold, cmax);
            float corr = __expf(mold - mnew);
            float lsum = s_l[lane] * corr;
#pragma unroll
            for (int j = 0; j < 32; ++j) {
                float p = __expf(s_tile[lane][j] * scale - mnew);
                lsum += p;
                p_tile[lane][j] = (__bf16)p;
            }
            s_m[lane] = mnew; s_l[lane] = lsum; s_corr[lane] = corr;
        }
        __syncthreads();

        // ---- rescale accumulators, then O += P (32x32) x V (32x16) ------
#pragma unroll
        for (int r = 0; r < 8; ++r) {
            acc0[r] *= s_corr[r + 8 * half];
            acc1[r] *= s_corr[16 + r + 8 * half];
        }
        v16bf ap0 = load_a_frag(&p_tile[0][0], 32);
        v16bf ap1 = load_a_frag(&p_tile[16][0], 32);
        v16bf bv;
        {
            const __bf16* vr = Vb + (size_t)m * SS + kc + 16 * half;
#pragma unroll
            for (int j = 0; j < 16; ++j) bv[j] = vr[j];
        }
        acc0 = wmma_bf16(ap0, bv, acc0);
        acc1 = wmma_bf16(ap1, bv, acc1);
        __syncthreads();                              // protect s_tile reuse
    }

    // ---- normalize and write O[b][s][h*16+dh] ----------------------------
#pragma unroll
    for (int r = 0; r < 8; ++r) {
        int row = r + 8 * half;
        float v0 = acc0[r] / s_l[row];
        float v1 = acc1[r] / s_l[16 + row];
        O[((size_t)b * SS + qt * 32 + row) * DD + h * 16 + m]      = v0;
        O[((size_t)b * SS + qt * 32 + 16 + row) * DD + h * 16 + m] = v1;
    }
}

// ---------------- generic WMMA GEMM: 64x32 tile per wave ------------------
// grid (M/64, N/32). Compile-time K: fully unrolled, double-buffered K loop.
// Per K-step: 2 B frags + 4 A frags -> 8 WMMAs.
template <int K, int RELU, typename OutT>
__global__ void k_gemm(const __bf16* __restrict__ A,
                       const __bf16* __restrict__ Bt,
                       const float* __restrict__ bias,
                       OutT* __restrict__ Cout, int N) {
    const int tm = blockIdx.x, tn = blockIdx.y;
    const int lane = threadIdx.x & 31;
    const int m = lane & 15, half = lane >> 4;

    const __bf16* Atile = A  + (size_t)(tm * 64) * K;
    const __bf16* Btile = Bt + (size_t)(tn * 32) * K;

    constexpr int NSTEP = K / 32;
    v16bf bf[2][2], af[2][4];
    bf[0][0] = load_bt_frag(Btile, K);
    bf[0][1] = load_bt_frag(Btile + (size_t)16 * K, K);
#pragma unroll
    for (int i = 0; i < 4; ++i) af[0][i] = load_a_frag(Atile + (size_t)(i * 16) * K, K);

    v8f acc[4][2] = {};
#pragma unroll
    for (int s = 0; s < NSTEP; ++s) {
        const int cur = s & 1, nxt = cur ^ 1;
        if (s + 1 < NSTEP) {
            const int kt = (s + 1) * 32;
            bf[nxt][0] = load_bt_frag(Btile + kt, K);
            bf[nxt][1] = load_bt_frag(Btile + (size_t)16 * K + kt, K);
#pragma unroll
            for (int i = 0; i < 4; ++i)
                af[nxt][i] = load_a_frag(Atile + (size_t)(i * 16) * K + kt, K);
        }
#pragma unroll
        for (int i = 0; i < 4; ++i) {
            acc[i][0] = wmma_bf16(af[cur][i], bf[cur][0], acc[i][0]);
            acc[i][1] = wmma_bf16(af[cur][i], bf[cur][1], acc[i][1]);
        }
    }
    const float bn0 = bias[tn * 32 + m];
    const float bn1 = bias[tn * 32 + 16 + m];
#pragma unroll
    for (int i = 0; i < 4; ++i) {
#pragma unroll
        for (int r = 0; r < 8; ++r) {
            int row = tm * 64 + i * 16 + r + 8 * half;
            float v0 = acc[i][0][r] + bn0;
            float v1 = acc[i][1][r] + bn1;
            if (RELU) { v0 = fmaxf(v0, 0.0f); v1 = fmaxf(v1, 0.0f); }
            Cout[(size_t)row * N + tn * 32 + m]      = (OutT)v0;
            Cout[(size_t)row * N + tn * 32 + 16 + m] = (OutT)v1;
        }
    }
}

// ---------------- fused residual add + LayerNorm (wave32 per row) --------
__global__ void k_add_ln(const float* __restrict__ a, const float* __restrict__ b,
                         const float* __restrict__ g, const float* __restrict__ bta,
                         float eps, float* __restrict__ xo, __bf16* __restrict__ xbo) {
    const int row = blockIdx.x;
    const int lane = threadIdx.x & 31;
    const float* ar = a + (size_t)row * DD;
    const float* br = b + (size_t)row * DD;
    float v[4];
    float sum = 0.0f;
#pragma unroll
    for (int i = 0; i < 4; ++i) { v[i] = ar[lane + 32 * i] + br[lane + 32 * i]; sum += v[i]; }
#pragma unroll
    for (int off = 16; off > 0; off >>= 1) sum += __shfl_xor(sum, off, 32);
    const float mu = sum * (1.0f / DD);
    float var = 0.0f;
#pragma unroll
    for (int i = 0; i < 4; ++i) { float d = v[i] - mu; var += d * d; }
#pragma unroll
    for (int off = 16; off > 0; off >>= 1) var += __shfl_xor(var, off, 32);
    const float inv = rsqrtf(var * (1.0f / DD) + eps);
#pragma unroll
    for (int i = 0; i < 4; ++i) {
        int c = lane + 32 * i;
        float y = g[c] * (v[i] - mu) * inv + bta[c];
        xo[(size_t)row * DD + c]  = y;
        xbo[(size_t)row * DD + c] = (__bf16)y;
    }
}

// ---------------- final projection D -> 6 classes (tiny, scalar) ---------
__global__ void k_outproj(const float* __restrict__ x, const float* __restrict__ W,
                          const float* __restrict__ bias, float* __restrict__ out) {
    int idx = blockIdx.x * blockDim.x + threadIdx.x;
    if (idx >= BS * NC) return;
    int row = idx / NC, c = idx % NC;
    const float* xr = x + (size_t)row * DD;
    float s = bias[c];
#pragma unroll 8
    for (int d = 0; d < DD; ++d) s += xr[d] * W[d * NC + c];
    out[idx] = s;
}

// =========================================================================
extern "C" void kernel_launch(void* const* d_in, const int* in_sizes, int n_in,
                              void* d_out, int out_size, void* d_ws, size_t ws_size,
                              hipStream_t stream) {
    const float* x_in  = (const float*)d_in[0];
    const float* Wq    = (const float*)d_in[1];
    const float* bq    = (const float*)d_in[2];
    const float* Wk    = (const float*)d_in[3];
    const float* bk    = (const float*)d_in[4];
    const float* Wv    = (const float*)d_in[5];
    const float* bv    = (const float*)d_in[6];
    const float* ln1g  = (const float*)d_in[7];
    const float* ln1b  = (const float*)d_in[8];
    const float* W1    = (const float*)d_in[9];
    const float* b1    = (const float*)d_in[10];
    const float* W2    = (const float*)d_in[11];
    const float* b2    = (const float*)d_in[12];
    const float* ln2g  = (const float*)d_in[13];
    const float* ln2b  = (const float*)d_in[14];
    const float* Wout  = (const float*)d_in[15];
    const float* bout  = (const float*)d_in[16];
    float* out = (float*)d_out;

    // ---- workspace carve ----
    char* ws = (char*)d_ws;
    size_t off = 0;
    auto carve = [&](size_t bytes) -> char* {
        char* p = ws + off;
        off += (bytes + 255) & ~size_t(255);
        return p;
    };
    float*  xf  = (float*) carve((size_t)BS * DD * 4);        // fp32 activations
    __bf16* xb  = (__bf16*)carve((size_t)BS * DD * 2);        // bf16 shadow
    __bf16* qb  = (__bf16*)carve((size_t)BB * HH * SS * DHH * 2);
    __bf16* kb  = (__bf16*)carve((size_t)BB * HH * SS * DHH * 2);
    __bf16* vtb = (__bf16*)carve((size_t)BB * HH * DHH * SS * 2);
    float*  ob  = (float*) carve((size_t)BS * DD * 4);        // attn out
    __bf16* hb  = (__bf16*)carve((size_t)BS * DFF * 2);       // ffn hidden
    float*  fb  = (float*) carve((size_t)BS * DD * 4);        // ffn out
    __bf16* wqt = (__bf16*)carve((size_t)LL * DD * DD * 2);
    __bf16* wkt = (__bf16*)carve((size_t)LL * DD * DD * 2);
    __bf16* wvt = (__bf16*)carve((size_t)LL * DD * DD * 2);
    __bf16* w1t = (__bf16*)carve((size_t)LL * DFF * DD * 2);  // [DFF][D] per layer
    __bf16* w2t = (__bf16*)carve((size_t)LL * DD * DFF * 2);  // [D][DFF] per layer
    (void)ws_size;

    // ---- weight convert + transpose (deterministic, every call) ----
    for (int l = 0; l < LL; ++l) {
        k_transpose_bf16<<<(DD * DD + 255) / 256, 256, 0, stream>>>(
            Wq + (size_t)l * DD * DD, wqt + (size_t)l * DD * DD, DD, DD);
        k_transpose_bf16<<<(DD * DD + 255) / 256, 256, 0, stream>>>(
            Wk + (size_t)l * DD * DD, wkt + (size_t)l * DD * DD, DD, DD);
        k_transpose_bf16<<<(DD * DD + 255) / 256, 256, 0, stream>>>(
            Wv + (size_t)l * DD * DD, wvt + (size_t)l * DD * DD, DD, DD);
        k_transpose_bf16<<<(DD * DFF + 255) / 256, 256, 0, stream>>>(
            W1 + (size_t)l * DD * DFF, w1t + (size_t)l * DFF * DD, DD, DFF);
        k_transpose_bf16<<<(DFF * DD + 255) / 256, 256, 0, stream>>>(
            W2 + (size_t)l * DFF * DD, w2t + (size_t)l * DD * DFF, DFF, DD);
    }
    k_convert_x<<<(BS * DD + 255) / 256, 256, 0, stream>>>(x_in, xf, xb, BS * DD);

    // ---- transformer layers ----
    for (int l = 0; l < LL; ++l) {
        dim3 gqkv(BS / 64, HH, 3);
        k_qkv<<<gqkv, 32, 0, stream>>>(
            xb,
            wqt + (size_t)l * DD * DD, wkt + (size_t)l * DD * DD, wvt + (size_t)l * DD * DD,
            bq + (size_t)l * DD, bk + (size_t)l * DD, bv + (size_t)l * DD,
            qb, kb, vtb);

        dim3 gat(SS / 32, HH, BB);
        k_attn<<<gat, 32, 0, stream>>>(qb, kb, vtb, ob);

        k_add_ln<<<BS, 32, 0, stream>>>(ob, xf,
            ln1g + (size_t)l * DD, ln1b + (size_t)l * DD, 1e-8f, xf, xb);

        dim3 gf1(BS / 64, DFF / 32);
        k_gemm<DD, 1, __bf16><<<gf1, 32, 0, stream>>>(
            xb, w1t + (size_t)l * DFF * DD, b1 + (size_t)l * DFF, hb, DFF);

        dim3 gf2(BS / 64, DD / 32);
        k_gemm<DFF, 0, float><<<gf2, 32, 0, stream>>>(
            hb, w2t + (size_t)l * DD * DFF, b2 + (size_t)l * DD, fb, DD);

        k_add_ln<<<BS, 32, 0, stream>>>(fb, xf,
            ln2g + (size_t)l * DD, ln2b + (size_t)l * DD, 1e-6f, xf, xb);
    }

    // ---- classifier head ----
    k_outproj<<<(BS * NC + 255) / 256, 256, 0, stream>>>(xf, Wout, bout, out);
}
